// Informer_80625126081029
// MI455X (gfx1250) — matrix operand order repaired and checked by reading, not compiled, not run on hardware
//
#include <hip/hip_runtime.h>
#include <hip/hip_bf16.h>
#include <math.h>

// ---------------------------------------------------------------- constants
#define BB    4
#define LL    2048
#define CIN   7
#define DM    512
#define NH    8
#define HD    64
#define DFF   2048
#define NTOP  40
#define MROWS (BB * LL)   // 8192
#define PRED  96

typedef _Float16 v16h __attribute__((ext_vector_type(16)));
typedef _Float16 v8h  __attribute__((ext_vector_type(8)));
typedef float    v8f  __attribute__((ext_vector_type(8)));

// ---------------------------------------------------------------- helpers
__device__ __forceinline__ float gelu_tanh(float x) {
    float x3 = x * x * x;
    return 0.5f * x * (1.0f + tanhf(0.7978845608028654f * (x + 0.044715f * x3)));
}

// ---------------------------------------------------------------- weight conversion with WMMA-B swizzle
// Per (k-tile kt, col-tile ct): 512-half chunk, [lane][16] so a lane's 16
// halves are one contiguous 32B v16h = its exact WMMA B fragment.
__global__ __launch_bounds__(256) void cvt_swz_b_kernel(const float* __restrict__ src,
                                                        _Float16* __restrict__ dst,
                                                        int N, int K) {
    int i = blockIdx.x * blockDim.x + threadIdx.x;   // over K*N
    if (i >= K * N) return;
    int chunk  = i >> 9;
    int within = i & 511;
    int lane   = within >> 4;
    int j      = within & 15;
    int nT     = N >> 4;
    int kt     = chunk / nT;
    int ct     = chunk - kt * nT;
    int k      = kt * 32 + (lane >> 4) * 16 + j;
    int n      = ct * 16 + (lane & 15);
    dst[i] = (_Float16)src[(size_t)k * N + n];
}

// ---------------------------------------------------------------- token embed + PE
__global__ __launch_bounds__(256) void embed_kernel(const float* __restrict__ x,
                                                    const float* __restrict__ w,
                                                    float* __restrict__ X) {
    int i = blockIdx.x * blockDim.x + threadIdx.x;   // MROWS*DM
    if (i >= MROWS * DM) return;
    int d  = i & (DM - 1);
    int bl = i >> 9;
    int l  = bl & (LL - 1);
    int b  = bl >> 11;
    float s = 0.0f;
#pragma unroll
    for (int kh = 0; kh < 3; ++kh) {
        int li = l + kh - 1;
        li = (li < 0) ? (LL - 1) : ((li >= LL) ? 0 : li);
        const float* xr = x + (size_t)(b * LL + li) * CIN;
#pragma unroll
        for (int c = 0; c < CIN; ++c)
            s += xr[c] * w[(kh * CIN + c) * DM + d];
    }
    int   ii   = d >> 1;
    float freq = __expf(-(float)(2 * ii) * (9.210340371976184f / (float)DM));
    float ang  = (float)l * freq;
    s += (d & 1) ? cosf(ang) : sinf(ang);
    X[i] = s;
}

// ---------------------------------------------------------------- WMMA GEMM
// C[M,N] = A[M,K](f32) * Bswz(f16, fragment-major) + bias ; optional GELU.
// 256 threads = 8 waves; block tile 128x128x32; wave tile 32x64 (2x4 frags).
// A double-buffered in LDS fragment-major: 2 x [miTile(8)][lane(32)][16] = 16KB.
__global__ __launch_bounds__(256) void wmma_gemm_kernel(const float* __restrict__ A,
                                                        const _Float16* __restrict__ Bswz,
                                                        const float* __restrict__ bias,
                                                        float* __restrict__ C,
                                                        int N, int K, int fuse_gelu) {
    __shared__ __align__(32) _Float16 AsH[2][8 * 32 * 16];   // 2 x 4096 halves

    const int t    = threadIdx.x;
    const int lane = t & 31;
    const int wave = t >> 5;
    const int wm   = wave & 3;                 // M offset 32*wm
    const int wn   = wave >> 2;                // N offset 64*wn
    const int m0   = blockIdx.y * 128;
    const int nT   = N >> 4;
    const int ct0  = blockIdx.x * 8 + wn * 4;  // this wave's first col-tile
    const int r    = lane & 15;
    const int hi   = lane >> 4;

    // stage one 128x32 A tile into fragment-major LDS buffer `buf`
    auto stageA = [&](int kt, int buf) {
#pragma unroll
        for (int it = 0; it < 2; ++it) {
            int s    = t + it * 256;
            int mi16 = s >> 6;
            int rem  = s & 63;
            int wl   = rem >> 1;               // lane slot 0..31
            int jh   = rem & 1;                // which 8-half of the fragment
            int grow = m0 + mi16 * 16 + (wl & 15);
            int gk   = kt * 32 + (wl >> 4) * 16 + jh * 8;
            const float4* g4 = (const float4*)(A + (size_t)grow * K + gk);
            float4 a0 = g4[0], a1 = g4[1];
            v8h tmp;
            tmp[0] = (_Float16)a0.x; tmp[1] = (_Float16)a0.y;
            tmp[2] = (_Float16)a0.z; tmp[3] = (_Float16)a0.w;
            tmp[4] = (_Float16)a1.x; tmp[5] = (_Float16)a1.y;
            tmp[6] = (_Float16)a1.z; tmp[7] = (_Float16)a1.w;
            *(v8h*)&AsH[buf][mi16 * 512 + wl * 16 + jh * 8] = tmp;
        }
    };

    v8f acc[2][4];
#pragma unroll
    for (int mi = 0; mi < 2; ++mi)
#pragma unroll
        for (int ni = 0; ni < 4; ++ni)
#pragma unroll
            for (int e = 0; e < 8; ++e) acc[mi][ni][e] = 0.0f;

    const int KT = K >> 5;
    stageA(0, 0);
    __syncthreads();

    for (int kt = 0; kt < KT; ++kt) {
        const int cur = kt & 1;
        // stage next tile into the other buffer (overlaps with compute below;
        // the end-of-iteration barrier keeps buffers race-free)
        if (kt + 1 < KT) {
            stageA(kt + 1, cur ^ 1);
            __builtin_prefetch(A + (size_t)(m0 + (t >> 1)) * K + (kt + 2) * 32, 0, 3);
        }

        // A fragments from LDS (one aligned 32B load each)
        v16h af0 = *(const v16h*)&AsH[cur][(wm * 2 + 0) * 512 + lane * 16];
        v16h af1 = *(const v16h*)&AsH[cur][(wm * 2 + 1) * 512 + lane * 16];

        // all four B fragments up-front in distinct registers so the eight
        // WMMAs issue back-to-back behind a single loadcnt wait
        const _Float16* bbase = Bswz + ((size_t)kt * nT + ct0) * 512;
        v16h bf0 = *(const v16h*)(bbase + 0 * 512 + lane * 16);
        v16h bf1 = *(const v16h*)(bbase + 1 * 512 + lane * 16);
        v16h bf2 = *(const v16h*)(bbase + 2 * 512 + lane * 16);
        v16h bf3 = *(const v16h*)(bbase + 3 * 512 + lane * 16);

        acc[0][0] = __builtin_amdgcn_wmma_f32_16x16x32_f16(false, af0, false, bf0, (short)0, acc[0][0], false, false);
        acc[1][0] = __builtin_amdgcn_wmma_f32_16x16x32_f16(false, af1, false, bf0, (short)0, acc[1][0], false, false);
        acc[0][1] = __builtin_amdgcn_wmma_f32_16x16x32_f16(false, af0, false, bf1, (short)0, acc[0][1], false, false);
        acc[1][1] = __builtin_amdgcn_wmma_f32_16x16x32_f16(false, af1, false, bf1, (short)0, acc[1][1], false, false);
        acc[0][2] = __builtin_amdgcn_wmma_f32_16x16x32_f16(false, af0, false, bf2, (short)0, acc[0][2], false, false);
        acc[1][2] = __builtin_amdgcn_wmma_f32_16x16x32_f16(false, af1, false, bf2, (short)0, acc[1][2], false, false);
        acc[0][3] = __builtin_amdgcn_wmma_f32_16x16x32_f16(false, af0, false, bf3, (short)0, acc[0][3], false, false);
        acc[1][3] = __builtin_amdgcn_wmma_f32_16x16x32_f16(false, af1, false, bf3, (short)0, acc[1][3], false, false);

        __syncthreads();
    }

    // epilogue: D layout VGPR rr -> M = rr + 8*hi ; col = lane&15
#pragma unroll
    for (int mi = 0; mi < 2; ++mi)
#pragma unroll
        for (int ni = 0; ni < 4; ++ni) {
            int   col = blockIdx.x * 128 + wn * 64 + ni * 16 + r;
            float bv  = bias ? bias[col] : 0.0f;
#pragma unroll
            for (int rr = 0; rr < 8; ++rr) {
                int   row = m0 + wm * 32 + mi * 16 + rr + 8 * hi;
                float v   = acc[mi][ni][rr] + bv;
                if (fuse_gelu) v = gelu_tanh(v);
                C[(size_t)row * N + col] = v;
            }
        }
}

// ---------------------------------------------------------------- residual + LayerNorm (512 cols)
__global__ __launch_bounds__(256) void ln_kernel(const float* __restrict__ x,
                                                 const float* __restrict__ res,
                                                 const float* __restrict__ g,
                                                 const float* __restrict__ bprm,
                                                 float* __restrict__ out) {
    __shared__ float red[256];
    int    tid  = threadIdx.x;
    size_t base = (size_t)blockIdx.x * DM;
    float  v0 = x[base + tid], v1 = x[base + 256 + tid];
    if (res) { v0 += res[base + tid]; v1 += res[base + 256 + tid]; }
    red[tid] = v0 + v1;
    __syncthreads();
    for (int s = 128; s > 0; s >>= 1) { if (tid < s) red[tid] += red[tid + s]; __syncthreads(); }
    float mean = red[0] * (1.0f / DM);
    __syncthreads();
    float d0 = v0 - mean, d1 = v1 - mean;
    red[tid] = d0 * d0 + d1 * d1;
    __syncthreads();
    for (int s = 128; s > 0; s >>= 1) { if (tid < s) red[tid] += red[tid + s]; __syncthreads(); }
    float inv = rsqrtf(red[0] * (1.0f / DM) + 1e-5f);
    out[base + tid]       = d0 * inv * g[tid]       + bprm[tid];
    out[base + 256 + tid] = d1 * inv * g[256 + tid] + bprm[256 + tid];
}

// ---------------------------------------------------------------- ProbSparse pieces
__global__ __launch_bounds__(256) void sampled_m_kernel(const float* __restrict__ Q,
                                                        const float* __restrict__ Kb,
                                                        float* __restrict__ Mout,
                                                        int layer) {
    int gid = blockIdx.x * blockDim.x + threadIdx.x;   // BB*NH*LL = 65536
    if (gid >= BB * NH * LL) return;
    int l  = gid & (LL - 1);
    int bh = gid >> 11;
    int b  = bh >> 3, h = bh & 7;
    const float* q = Q + (size_t)(b * LL + l) * DM + h * HD;
    float mx = -3.4e38f, sum = 0.0f;
    for (int u = 0; u < NTOP; ++u) {
        unsigned v = (unsigned)l * 1103515245u + (unsigned)u * 2654435761u +
                     (unsigned)layer * 40503u + 12345u;
        v ^= v >> 13; v *= 0x9e3779b9u; v ^= v >> 16;
        int kk = (int)(v & (LL - 1));
        const float* kr = Kb + (size_t)(b * LL + kk) * DM + h * HD;
        float d = 0.0f;
#pragma unroll
        for (int j = 0; j < HD; ++j) d += q[j] * kr[j];
        mx = fmaxf(mx, d);
        sum += d;
    }
    Mout[gid] = mx - sum * (1.0f / (float)LL);
}

__global__ __launch_bounds__(256) void topk_kernel(float* __restrict__ Mwork,
                                                   int* __restrict__ mtop) {
    __shared__ float sv[256];
    __shared__ int   si[256];
    int    tid  = threadIdx.x;
    float* Mrow = Mwork + (size_t)blockIdx.x * LL;
    for (int sel = 0; sel < NTOP; ++sel) {
        float best = -3.4e38f; int bi = 0;
        for (int l = tid; l < LL; l += 256) {
            float v = Mrow[l];
            if (v > best) { best = v; bi = l; }
        }
        sv[tid] = best; si[tid] = bi;
        __syncthreads();
        for (int s = 128; s > 0; s >>= 1) {
            if (tid < s && sv[tid + s] > sv[tid]) { sv[tid] = sv[tid + s]; si[tid] = si[tid + s]; }
            __syncthreads();
        }
        if (tid == 0) { mtop[blockIdx.x * NTOP + sel] = si[0]; Mrow[si[0]] = -3.4e38f; }
        __syncthreads();
    }
}

__global__ void vmean_kernel(const float* __restrict__ V, float* __restrict__ vmean) {
    int bh = blockIdx.x, d = threadIdx.x;       // 32 blocks x 64 threads
    int b = bh >> 3, h = bh & 7;
    float s = 0.0f;
    for (int l = 0; l < LL; ++l) s += V[(size_t)(b * LL + l) * DM + h * HD + d];
    vmean[bh * HD + d] = s * (1.0f / (float)LL);
}

__global__ __launch_bounds__(256) void fillctx_kernel(const float* __restrict__ vmean,
                                                      float* __restrict__ ctx) {
    int i = blockIdx.x * blockDim.x + threadIdx.x;  // MROWS*DM
    if (i >= MROWS * DM) return;
    int c = i & (DM - 1);
    int b = i >> 20;
    int h = c >> 6, d = c & 63;
    ctx[i] = vmean[(b * NH + h) * HD + d];
}

__global__ __launch_bounds__(256) void topctx_kernel(const float* __restrict__ Q,
                                                     const float* __restrict__ Kb,
                                                     const float* __restrict__ V,
                                                     const int* __restrict__ mtop,
                                                     float* __restrict__ ctx) {
    __shared__ float sc[LL];
    __shared__ float qs[HD];
    __shared__ float red[256];
    int tid = threadIdx.x;
    int u   = blockIdx.x % NTOP;
    int bh  = blockIdx.x / NTOP;
    int b   = bh >> 3, h = bh & 7;
    int lq  = mtop[bh * NTOP + u];
    if (tid < HD) qs[tid] = Q[(size_t)(b * LL + lq) * DM + h * HD + tid];
    __syncthreads();

    float lmax = -3.4e38f;
    for (int l = tid; l < LL; l += 256) {
        const float* kr = Kb + (size_t)(b * LL + l) * DM + h * HD;
        float d = 0.0f;
#pragma unroll
        for (int j = 0; j < HD; ++j) d += qs[j] * kr[j];
        d *= 0.125f;
        sc[l] = d;
        lmax = fmaxf(lmax, d);
    }
    red[tid] = lmax;
    __syncthreads();
    for (int s = 128; s > 0; s >>= 1) { if (tid < s) red[tid] = fmaxf(red[tid], red[tid + s]); __syncthreads(); }
    float mx = red[0];
    __syncthreads();
    float lsum = 0.0f;
    for (int l = tid; l < LL; l += 256) { float e = __expf(sc[l] - mx); sc[l] = e; lsum += e; }
    red[tid] = lsum;
    __syncthreads();
    for (int s = 128; s > 0; s >>= 1) { if (tid < s) red[tid] += red[tid + s]; __syncthreads(); }
    float inv = 1.0f / red[0];
    __syncthreads();
    if (tid < HD) {
        float acc = 0.0f;
        for (int l = 0; l < LL; ++l) acc += sc[l] * V[(size_t)(b * LL + l) * DM + h * HD + tid];
        ctx[(size_t)(b * LL + lq) * DM + h * HD + tid] = acc * inv;
    }
}

// ---------------------------------------------------------------- final 512->1 proj
__global__ void final_proj_kernel(const float* __restrict__ X,
                                  const float* __restrict__ pw,
                                  const float* __restrict__ pb,
                                  float* __restrict__ out) {
    int i = blockIdx.x * blockDim.x + threadIdx.x;   // BB*PRED = 384
    if (i >= BB * PRED) return;
    int b = i / PRED, tpos = i % PRED;
    size_t row = (size_t)(b * LL + (LL - PRED + tpos)) * DM;
    float s = pb[0];
#pragma unroll 8
    for (int k = 0; k < DM; ++k) s += X[row + k] * pw[k];
    out[i] = s;
}

// ================================================================ launch
struct LayerPtrs {
    const float *wq, *bq, *wk, *bk, *wv, *bv, *wo, *bo;
    const float *w1, *bf1, *w2, *bf2, *g1, *b1, *g2, *b2;
};

extern "C" void kernel_launch(void* const* d_in, const int* in_sizes, int n_in,
                              void* d_out, int out_size, void* d_ws, size_t ws_size,
                              hipStream_t stream) {
    (void)out_size; (void)ws_size;
    const float *x_enc, *token_w, *norm_g, *norm_b, *proj_w, *proj_b;
    LayerPtrs lp[2];

    if (n_in >= 38 && in_sizes[0] == BB * LL * CIN) {
        // dict-insertion order
        x_enc   = (const float*)d_in[0];
        token_w = (const float*)d_in[1];
        for (int i = 0; i < 2; ++i) {
            int o = 2 + i * 16;
            lp[i].wq  = (const float*)d_in[o + 0];  lp[i].bq  = (const float*)d_in[o + 1];
            lp[i].wk  = (const float*)d_in[o + 2];  lp[i].bk  = (const float*)d_in[o + 3];
            lp[i].wv  = (const float*)d_in[o + 4];  lp[i].bv  = (const float*)d_in[o + 5];
            lp[i].wo  = (const float*)d_in[o + 6];  lp[i].bo  = (const float*)d_in[o + 7];
            lp[i].w1  = (const float*)d_in[o + 8];  lp[i].bf1 = (const float*)d_in[o + 9];
            lp[i].w2  = (const float*)d_in[o + 10]; lp[i].bf2 = (const float*)d_in[o + 11];
            lp[i].g1  = (const float*)d_in[o + 12]; lp[i].b1  = (const float*)d_in[o + 13];
            lp[i].g2  = (const float*)d_in[o + 14]; lp[i].b2  = (const float*)d_in[o + 15];
        }
        norm_g = (const float*)d_in[34]; norm_b = (const float*)d_in[35];
        proj_w = (const float*)d_in[36]; proj_b = (const float*)d_in[37];
    } else {
        // jax pytree sorted-keys order
        for (int i = 0; i < 2; ++i) {
            int o = i * 16;
            lp[i].b1  = (const float*)d_in[o + 0];  lp[i].b2  = (const float*)d_in[o + 1];
            lp[i].bf1 = (const float*)d_in[o + 2];  lp[i].bf2 = (const float*)d_in[o + 3];
            lp[i].bk  = (const float*)d_in[o + 4];  lp[i].bo  = (const float*)d_in[o + 5];
            lp[i].bq  = (const float*)d_in[o + 6];  lp[i].bv  = (const float*)d_in[o + 7];
            lp[i].g1  = (const float*)d_in[o + 8];  lp[i].g2  = (const float*)d_in[o + 9];
            lp[i].w1  = (const float*)d_in[o + 10]; lp[i].w2  = (const float*)d_in[o + 11];
            lp[i].wk  = (const float*)d_in[o + 12]; lp[i].wo  = (const float*)d_in[o + 13];
            lp[i].wq  = (const float*)d_in[o + 14]; lp[i].wv  = (const float*)d_in[o + 15];
        }
        norm_b  = (const float*)d_in[32]; norm_g  = (const float*)d_in[33];
        proj_b  = (const float*)d_in[34]; proj_w  = (const float*)d_in[35];
        token_w = (const float*)d_in[36]; x_enc   = (const float*)d_in[37];
    }

    // -------- workspace carve-out
    size_t off = 0;
    auto carve = [&](size_t bytes) -> void* {
        void* p = (char*)d_ws + off;
        off += (bytes + 255) & ~(size_t)255;
        return p;
    };
    float* X    = (float*)carve((size_t)MROWS * DM * 4);
    float* Qb   = (float*)carve((size_t)MROWS * DM * 4);
    float* Kb   = (float*)carve((size_t)MROWS * DM * 4);
    float* Vb   = (float*)carve((size_t)MROWS * DM * 4);
    float* CTX  = (float*)carve((size_t)MROWS * DM * 4);
    float* ATT  = (float*)carve((size_t)MROWS * DM * 4);   // attn_out, reused as y2
    float* Y1   = (float*)carve((size_t)MROWS * DFF * 4);
    float* Mm   = (float*)carve((size_t)BB * NH * LL * 4);
    int*   mtop = (int*)  carve((size_t)BB * NH * NTOP * 4);
    float* vmn  = (float*)carve((size_t)BB * NH * HD * 4);
    _Float16* Wh[2][6];
    const int wsz[6] = { DM * DM, DM * DM, DM * DM, DM * DM, DM * DFF, DFF * DM };
    const int wN [6] = { DM, DM, DM, DM, DFF, DM };
    const int wK [6] = { DM, DM, DM, DM, DM, DFF };
    for (int i = 0; i < 2; ++i)
        for (int j = 0; j < 6; ++j)
            Wh[i][j] = (_Float16*)carve((size_t)wsz[j] * 2);

    // -------- weight conversion with fragment swizzle (f32 -> f16)
    for (int i = 0; i < 2; ++i) {
        const float* src[6] = { lp[i].wq, lp[i].wk, lp[i].wv, lp[i].wo, lp[i].w1, lp[i].w2 };
        for (int j = 0; j < 6; ++j)
            cvt_swz_b_kernel<<<(wsz[j] + 255) / 256, 256, 0, stream>>>(src[j], Wh[i][j],
                                                                       wN[j], wK[j]);
    }

    // -------- embed + positional encoding
    embed_kernel<<<(MROWS * DM) / 256, 256, 0, stream>>>(x_enc, token_w, X);

    dim3 g512(DM / 128, MROWS / 128);     // (4, 64)
    dim3 gFF (DFF / 128, MROWS / 128);    // (16, 64)

    for (int i = 0; i < 2; ++i) {
        wmma_gemm_kernel<<<g512, 256, 0, stream>>>(X, Wh[i][0], lp[i].bq, Qb, DM, DM, 0);
        wmma_gemm_kernel<<<g512, 256, 0, stream>>>(X, Wh[i][1], lp[i].bk, Kb, DM, DM, 0);
        wmma_gemm_kernel<<<g512, 256, 0, stream>>>(X, Wh[i][2], lp[i].bv, Vb, DM, DM, 0);
        sampled_m_kernel<<<(BB * NH * LL) / 256, 256, 0, stream>>>(Qb, Kb, Mm, i);
        topk_kernel<<<BB * NH, 256, 0, stream>>>(Mm, mtop);
        vmean_kernel<<<BB * NH, HD, 0, stream>>>(Vb, vmn);
        fillctx_kernel<<<(MROWS * DM) / 256, 256, 0, stream>>>(vmn, CTX);
        topctx_kernel<<<BB * NH * NTOP, 256, 0, stream>>>(Qb, Kb, Vb, mtop, CTX);
        wmma_gemm_kernel<<<g512, 256, 0, stream>>>(CTX, Wh[i][3], lp[i].bo, ATT, DM, DM, 0);
        ln_kernel<<<MROWS, 256, 0, stream>>>(X, ATT, lp[i].g1, lp[i].b1, X);
        wmma_gemm_kernel<<<gFF, 256, 0, stream>>>(X, Wh[i][4], lp[i].bf1, Y1, DFF, DM, 1);
        wmma_gemm_kernel<<<g512, 256, 0, stream>>>(Y1, Wh[i][5], lp[i].bf2, ATT, DM, DFF, 0);
        ln_kernel<<<MROWS, 256, 0, stream>>>(X, ATT, lp[i].g2, lp[i].b2, X);
    }

    ln_kernel<<<MROWS, 256, 0, stream>>>(X, nullptr, norm_g, norm_b, X);
    final_proj_kernel<<<(BB * PRED + 255) / 256, 256, 0, stream>>>(X, proj_w, proj_b, (float*)d_out);
}
